// MultiHeadAttention_12592844111922
// MI455X (gfx1250) — matrix operand order repaired and checked by reading, not compiled, not run on hardware
//
#include <hip/hip_runtime.h>

// ---------------------------------------------------------------------------
// MI455X (gfx1250) fused causal MHA forward.
// B=4, T=2048, C=1024, H=16, D=64.  All matmuls via v_wmma_f32_16x16x32_bf16.
// ---------------------------------------------------------------------------

typedef __attribute__((ext_vector_type(16))) __bf16 v16bf;
typedef __attribute__((ext_vector_type(8)))  __bf16 bfx8;
typedef __attribute__((ext_vector_type(4)))  __bf16 bfx4;
typedef __attribute__((ext_vector_type(8)))  float  v8f;

union FragA { v16bf v; bfx8 h[2]; };   // 16x32 bf16 A fragment (per lane)
union FragB { v16bf v; bfx8 h[2]; };   // 32x16 bf16 B fragment (per lane)

static __device__ __forceinline__
v8f wmma_bf16(v16bf a, v16bf b, v8f c) {
    // D = A(16x32) * B(32x16) + C, f32 accumulate
    return __builtin_amdgcn_wmma_f32_16x16x32_bf16(
        /*neg_a=*/false, a, /*neg_b=*/false, b,
        /*c_mod=*/(short)0, c, /*reuse_a=*/false, /*reuse_b=*/false);
}

static constexpr int Bb = 4;
static constexpr int Ts = 2048;
static constexpr int Cc = 1024;
static constexpr int Hh = 16;
static constexpr int Dd = 64;
static constexpr int Mtot = Bb * Ts;        // 8192 rows

// ---------------------------------------------------------------------------
// Kernel 0a: fp32 -> bf16 streaming convert (x)
// ---------------------------------------------------------------------------
__global__ __launch_bounds__(256) void k_convert_f32_bf16(
    const float* __restrict__ src, __bf16* __restrict__ dst, int n) {
    int i = (blockIdx.x * 256 + threadIdx.x) * 4;
    if (i + 3 < n) {
        float4 f = *(const float4*)(src + i);
        bfx4 o;
        o[0] = (__bf16)f.x; o[1] = (__bf16)f.y;
        o[2] = (__bf16)f.z; o[3] = (__bf16)f.w;
        *(bfx4*)(dst + i) = o;
    }
}

// ---------------------------------------------------------------------------
// Kernel 0b: fp32 [K,N] -> bf16 transposed [N,K] (weights), coalesced read
// ---------------------------------------------------------------------------
__global__ __launch_bounds__(256) void k_transpose_w_bf16(
    const float* __restrict__ W, __bf16* __restrict__ WT, int K, int N) {
    int n = blockIdx.x * 256 + threadIdx.x;
    int k = blockIdx.y;
    if (n < N) WT[(size_t)n * K + k] = (__bf16)W[(size_t)k * N + n];
}

// ---------------------------------------------------------------------------
// Kernel 1: QKV GEMM.  A = x_bf16 [8192,1024], BT = WqkvT [3072,1024].
// Each wave computes a 64x32 tile (4 M-subtiles x 2 N-subtiles of 16x16).
// Operands streamed straight from global (L2-resident), no LDS.
// Output scattered into Q [B,H,T,D], K [B,H,T,D], V^T [B,H,D,T] (bf16).
// ---------------------------------------------------------------------------
__global__ __launch_bounds__(256) void k_qkv_gemm(
    const __bf16* __restrict__ A, const __bf16* __restrict__ BT,
    __bf16* __restrict__ qws, __bf16* __restrict__ kws,
    __bf16* __restrict__ vtws) {
    const int K = Cc;
    const int lane   = threadIdx.x & 31;
    const int w      = threadIdx.x >> 5;
    const int laneLo = lane & 15;
    const int laneHi = lane >> 4;
    const int m0 = blockIdx.y * 128 + (w >> 2) * 64;
    const int n0 = blockIdx.x * 128 + (w & 3) * 32;

    v8f acc[4][2] = {};
    for (int kb = 0; kb < K; kb += 32) {
        FragA a[4];
        FragB b[2];
#pragma unroll
        for (int mt = 0; mt < 4; ++mt) {
            const __bf16* pa = A + (size_t)(m0 + mt * 16 + laneLo) * K + kb + laneHi * 16;
            a[mt].h[0] = *(const bfx8*)(pa);
            a[mt].h[1] = *(const bfx8*)(pa + 8);
        }
#pragma unroll
        for (int nt = 0; nt < 2; ++nt) {
            b[nt].v = *(const v16bf*)(BT + (size_t)(n0 + nt * 16 + laneLo) * K
                                         + kb + laneHi * 16);
        }
        if (kb + 32 < K)
            __builtin_prefetch(A + (size_t)(m0 + laneLo) * K + kb + 32, 0, 1);
#pragma unroll
        for (int mt = 0; mt < 4; ++mt)
#pragma unroll
            for (int nt = 0; nt < 2; ++nt)
                acc[mt][nt] = wmma_bf16(a[mt].v, b[nt].v, acc[mt][nt]);
    }

    // Epilogue: wave-uniform routing (n-range of 32 never crosses q/k/v or head
    // boundaries).
    const int which = n0 / Cc;          // 0=q 1=k 2=v
    const int c0    = n0 % Cc;
    const int h     = c0 >> 6;
    const int dbase = c0 & 63;
#pragma unroll
    for (int mt = 0; mt < 4; ++mt) {
#pragma unroll
        for (int r = 0; r < 8; ++r) {
            const int m  = m0 + mt * 16 + r + 8 * laneHi;
            const int bi = m >> 11;          // / 2048
            const int t  = m & 2047;
            const int bh = bi * Hh + h;
#pragma unroll
            for (int nt = 0; nt < 2; ++nt) {
                const int d = dbase + nt * 16 + laneLo;
                const __bf16 val = (__bf16)acc[mt][nt][r];
                if (which == 0)      qws [((size_t)bh * Ts + t) * Dd + d] = val;
                else if (which == 1) kws [((size_t)bh * Ts + t) * Dd + d] = val;
                else                 vtws[((size_t)bh * Dd + d) * Ts + t] = val;
            }
        }
    }
}

// ---------------------------------------------------------------------------
// Kernel 2: flash attention (causal, online softmax).
// Block = 128 threads = 4 waves; wave handles 16 query rows of one (b,h).
// Per 32-key block: S = Q K^T (4 WMMAs), row-softmax via shfl reductions,
// P reshaped through LDS into an A fragment, O += P V (4 WMMAs).
// ---------------------------------------------------------------------------
__global__ __launch_bounds__(128) void k_flash_attn(
    const __bf16* __restrict__ q, const __bf16* __restrict__ k,
    const __bf16* __restrict__ vt, __bf16* __restrict__ attn) {
    __shared__ alignas(32) __bf16 ptile[4][16 * 32];

    const int lane   = threadIdx.x & 31;
    const int w      = threadIdx.x >> 5;
    const int laneLo = lane & 15;
    const int laneHi = lane >> 4;
    const int bh = blockIdx.y;
    const int b  = bh >> 4;
    const int h  = bh & 15;
    const int q0 = blockIdx.x * 64 + w * 16;

    const __bf16* qp  = q  + (size_t)bh * Ts * Dd;
    const __bf16* kp  = k  + (size_t)bh * Ts * Dd;
    const __bf16* vtp = vt + (size_t)bh * Dd * Ts;

    // Load Q fragments once (two K-steps of 32 over D=64).
    FragA aq[2];
#pragma unroll
    for (int ks = 0; ks < 2; ++ks) {
        const __bf16* p = qp + (size_t)(q0 + laneLo) * Dd + ks * 32 + laneHi * 16;
        aq[ks].h[0] = *(const bfx8*)(p);
        aq[ks].h[1] = *(const bfx8*)(p + 8);
    }

    float mrow[8], lrow[8];
    v8f o[4] = {};
#pragma unroll
    for (int r = 0; r < 8; ++r) { mrow[r] = -3.0e38f; lrow[r] = 0.0f; }

    const int kbmax = q0 / 32 + 1;   // causal bound (keys <= q0+15)
    for (int kb = 0; kb < kbmax; ++kb) {
        // ---- S = Q K^T (16 x 32 keys), f32 accum --------------------------
        v8f s[2] = {};
#pragma unroll
        for (int nt = 0; nt < 2; ++nt) {
#pragma unroll
            for (int ks = 0; ks < 2; ++ks) {
                FragB bk;
                bk.v = *(const v16bf*)(kp
                        + (size_t)(kb * 32 + nt * 16 + laneLo) * Dd
                        + ks * 32 + laneHi * 16);
                s[nt] = wmma_bf16(aq[ks].v, bk.v, s[nt]);
            }
        }
        // ---- scale + causal mask -----------------------------------------
        const int qbase = q0 + 8 * laneHi;
#pragma unroll
        for (int r = 0; r < 8; ++r) {
            const int qr = qbase + r;
#pragma unroll
            for (int nt = 0; nt < 2; ++nt) {
                const int kc = kb * 32 + nt * 16 + laneLo;
                float val = s[nt][r] * 0.125f;          // D^-0.5
                if (kc > qr) val = -3.0e38f;
                s[nt][r] = val;
            }
        }
        // ---- online softmax update ---------------------------------------
#pragma unroll
        for (int r = 0; r < 8; ++r) {
            float mx = fmaxf(s[0][r], s[1][r]);
            mx = fmaxf(mx, __shfl_xor(mx, 1));
            mx = fmaxf(mx, __shfl_xor(mx, 2));
            mx = fmaxf(mx, __shfl_xor(mx, 4));
            mx = fmaxf(mx, __shfl_xor(mx, 8));
            const float mnew = fmaxf(mrow[r], mx);
            const float corr = __expf(mrow[r] - mnew);
            mrow[r] = mnew;
            const float p0 = __expf(s[0][r] - mnew);
            const float p1 = __expf(s[1][r] - mnew);
            s[0][r] = p0; s[1][r] = p1;
            float rs = p0 + p1;
            rs += __shfl_xor(rs, 1);
            rs += __shfl_xor(rs, 2);
            rs += __shfl_xor(rs, 4);
            rs += __shfl_xor(rs, 8);
            lrow[r] = lrow[r] * corr + rs;
#pragma unroll
            for (int nt4 = 0; nt4 < 4; ++nt4) o[nt4][r] *= corr;
        }
        // ---- P (C-layout) -> LDS -> A fragment ---------------------------
        __bf16* pt = ptile[w];
#pragma unroll
        for (int r = 0; r < 8; ++r) {
            const int row = r + 8 * laneHi;
            pt[row * 32 + laneLo]      = (__bf16)s[0][r];
            pt[row * 32 + 16 + laneLo] = (__bf16)s[1][r];
        }
        asm volatile("s_wait_dscnt 0" ::: "memory");   // wave-local LDS RAW
        FragA pa;
        {
            const __bf16* pp = pt + laneLo * 32 + laneHi * 16;
            pa.h[0] = *(const bfx8*)(pp);
            pa.h[1] = *(const bfx8*)(pp + 8);
        }
        // ---- O += P * V  (V stored transposed -> contiguous B frags) -----
#pragma unroll
        for (int nt4 = 0; nt4 < 4; ++nt4) {
            FragB bv;
            bv.v = *(const v16bf*)(vtp
                    + (size_t)(nt4 * 16 + laneLo) * Ts
                    + kb * 32 + laneHi * 16);
            o[nt4] = wmma_bf16(pa.v, bv.v, o[nt4]);
        }
        asm volatile("s_wait_dscnt 0" ::: "memory");   // fence before re-store
    }

    // ---- normalize + store attn_out as [B,T,C] bf16 ----------------------
#pragma unroll
    for (int r = 0; r < 8; ++r) {
        const float inv = 1.0f / lrow[r];
        const int t = q0 + r + 8 * laneHi;
#pragma unroll
        for (int nt4 = 0; nt4 < 4; ++nt4) {
            const int d = nt4 * 16 + laneLo;
            attn[((size_t)b * Ts + t) * Cc + h * Dd + d] = (__bf16)(o[nt4][r] * inv);
        }
    }
}

// ---------------------------------------------------------------------------
// Kernel 3: output projection.  A = attn_bf16 [8192,1024], BT = WoutT, f32 out.
// ---------------------------------------------------------------------------
__global__ __launch_bounds__(256) void k_out_gemm(
    const __bf16* __restrict__ A, const __bf16* __restrict__ BT,
    float* __restrict__ out) {
    const int K = Cc;
    const int lane   = threadIdx.x & 31;
    const int w      = threadIdx.x >> 5;
    const int laneLo = lane & 15;
    const int laneHi = lane >> 4;
    const int m0 = blockIdx.y * 128 + (w >> 2) * 64;
    const int n0 = blockIdx.x * 128 + (w & 3) * 32;

    v8f acc[4][2] = {};
    for (int kb = 0; kb < K; kb += 32) {
        FragA a[4];
        FragB b[2];
#pragma unroll
        for (int mt = 0; mt < 4; ++mt) {
            const __bf16* pa = A + (size_t)(m0 + mt * 16 + laneLo) * K + kb + laneHi * 16;
            a[mt].h[0] = *(const bfx8*)(pa);
            a[mt].h[1] = *(const bfx8*)(pa + 8);
        }
#pragma unroll
        for (int nt = 0; nt < 2; ++nt) {
            b[nt].v = *(const v16bf*)(BT + (size_t)(n0 + nt * 16 + laneLo) * K
                                         + kb + laneHi * 16);
        }
#pragma unroll
        for (int mt = 0; mt < 4; ++mt)
#pragma unroll
            for (int nt = 0; nt < 2; ++nt)
                acc[mt][nt] = wmma_bf16(a[mt].v, b[nt].v, acc[mt][nt]);
    }
#pragma unroll
    for (int mt = 0; mt < 4; ++mt)
#pragma unroll
        for (int r = 0; r < 8; ++r) {
            const int m = m0 + mt * 16 + r + 8 * laneHi;
#pragma unroll
            for (int nt = 0; nt < 2; ++nt) {
                const int n = n0 + nt * 16 + laneLo;
                out[(size_t)m * Cc + n] = acc[mt][nt][r];
            }
        }
}

// ---------------------------------------------------------------------------
// Launcher.  Workspace layout (bytes):
//   [0,16M)   x_bf16        [8192,1024]
//   [16M,22M) WqkvT_bf16    [3072,1024]
//   [22M,24M) WoutT_bf16    [1024,1024]
//   [24M,40M) Q  bf16       [B,H,T,D]
//   [40M,56M) K  bf16       [B,H,T,D]
//   [56M,72M) V^T bf16      [B,H,D,T]
//   [72M,88M) attn bf16     [8192,1024]
// ---------------------------------------------------------------------------
extern "C" void kernel_launch(void* const* d_in, const int* in_sizes, int n_in,
                              void* d_out, int out_size, void* d_ws, size_t ws_size,
                              hipStream_t stream) {
    (void)in_sizes; (void)n_in; (void)out_size; (void)ws_size;
    const float* x    = (const float*)d_in[0];
    const float* Wqkv = (const float*)d_in[1];
    const float* Wout = (const float*)d_in[2];
    float* out = (float*)d_out;

    char* ws = (char*)d_ws;
    const size_t MB = 1024 * 1024;
    __bf16* xb    = (__bf16*)(ws + 0 * MB);
    __bf16* wqkvT = (__bf16*)(ws + 16 * MB);
    __bf16* woutT = (__bf16*)(ws + 22 * MB);
    __bf16* qws   = (__bf16*)(ws + 24 * MB);
    __bf16* kws   = (__bf16*)(ws + 40 * MB);
    __bf16* vtws  = (__bf16*)(ws + 56 * MB);
    __bf16* attnb = (__bf16*)(ws + 72 * MB);

    // 0) precision conversion + weight transpose (one streaming pass)
    k_convert_f32_bf16<<<(Mtot * Cc) / (256 * 4), 256, 0, stream>>>(x, xb, Mtot * Cc);
    k_transpose_w_bf16<<<dim3(3 * Cc / 256, Cc), 256, 0, stream>>>(Wqkv, wqkvT, Cc, 3 * Cc);
    k_transpose_w_bf16<<<dim3(Cc / 256, Cc), 256, 0, stream>>>(Wout, woutT, Cc, Cc);

    // 1) QKV projection (WMMA bf16), writes Q/K/V^T in attention-native layout
    k_qkv_gemm<<<dim3(3 * Cc / 128, Mtot / 128), 256, 0, stream>>>(
        xb, wqkvT, qws, kws, vtws);

    // 2) causal flash attention (WMMA bf16 + online softmax)
    k_flash_attn<<<dim3(Ts / 64, Bb * Hh), 128, 0, stream>>>(
        qws, kws, vtws, attnb);

    // 3) output projection (WMMA bf16, f32 store)
    k_out_gemm<<<dim3(Cc / 128, Mtot / 128), 256, 0, stream>>>(
        attnb, woutT, out);
}
